// SpikingLayer_68401649156434
// MI455X (gfx1250) — matrix operand order repaired and checked by reading, not compiled, not run on hardware
//
#include <hip/hip_runtime.h>

// ---------------------------------------------------------------------------
// SNN layer for MI455X (gfx1250):
//   pre :  X,W f32 -> bf16 in d_ws (memory-bound, removes VALU from GEMM)
//   gemm:  current = X[10240,2048] * W^T + b   (v_wmma_f32_16x16x32_bf16,
//          128x256 block tile, 64x64 per-wave tile, async Global->LDS fill)
//   scan:  LIF over T=20 in place on d_out     (memory-bound)
// Fallback (small ws): f32-source kernel converting bf16 during LDS fill.
// ---------------------------------------------------------------------------

typedef __attribute__((ext_vector_type(16))) __bf16        v16bf;
typedef __attribute__((ext_vector_type(8)))  __bf16        v8bf;
typedef __attribute__((ext_vector_type(8)))  float         v8f;
typedef __attribute__((ext_vector_type(8)))  unsigned short u16x8;
typedef __attribute__((ext_vector_type(4)))  float         f32x4;
typedef __attribute__((ext_vector_type(4)))  int           v4i;

typedef __attribute__((address_space(1))) v4i gv4i;   // global b128
typedef __attribute__((address_space(3))) v4i lv4i;   // LDS b128

#define TAU    0.5f
#define VTH    1.0f
#define TSTEPS 20
#define DIN    2048
#define DOUT   2048
#define MROWS  10240   // B*T = 512*20
#define NBATCH 512

#define BM  128
#define BNF 256                  // fast-kernel block N
#define BNS 128                  // fallback block N
#define BK  32
#define LROW 40                  // padded LDS row (halves): 16B-aligned, conflict-free
#define NSTAGE (DIN / BK)        // 64
#define PFDIST 4

#define XELEMS ((size_t)MROWS * DIN)          // 20,971,520
#define WELEMS ((size_t)DOUT * DIN)           //  4,194,304
#define WS_NEEDED ((XELEMS + WELEMS) * 2)     // ~50 MB of bf16

#if __has_builtin(__builtin_amdgcn_global_load_async_to_lds_b128)
#define HAVE_ASYNC_LDS 1
#else
#define HAVE_ASYNC_LDS 0
#endif

__device__ __forceinline__ unsigned short f32_to_bf16_bits(float f) {
  unsigned u = __builtin_bit_cast(unsigned, f);
  u += 0x7fffu + ((u >> 16) & 1u);    // round-to-nearest-even
  return (unsigned short)(u >> 16);
}

#if HAVE_ASYNC_LDS
// async copy of one 16B chunk: global -> LDS (per-lane addresses, ASYNCcnt)
__device__ __forceinline__ void async_cp16(const unsigned short* g,
                                           unsigned short* l) {
  gv4i* gp = (gv4i*)(v4i*)(void*)g;   // drop const, retype, addrspacecast
  lv4i* lp = (lv4i*)(v4i*)l;
  __builtin_amdgcn_global_load_async_to_lds_b128(gp, lp, 0, 0);
}
#endif

__device__ __forceinline__ v16bf lds_frag(const unsigned short* p0,
                                          const unsigned short* p1) {
  v8bf lo = *(const v8bf*)p0;
  v8bf hi = *(const v8bf*)p1;
  return __builtin_shufflevector(lo, hi, 0, 1, 2, 3, 4, 5, 6, 7,
                                        8, 9, 10, 11, 12, 13, 14, 15);
}

// ---- GEMM core: 4 x NT WMMAs per wave per k-stage -------------------------
template <int NT>
__device__ __forceinline__ void mma_stage(const unsigned short* sAst,
                                          const unsigned short* sBst,
                                          int wm, int wn, int l15, int hh,
                                          v8f (&acc)[4][NT]) {
  v16bf bfr[NT], afr[4];
#pragma unroll
  for (int nt = 0; nt < NT; ++nt) {
    const unsigned short* p =
        sBst + (wn * (16 * NT) + nt * 16 + l15) * LROW + hh * 16;
    bfr[nt] = lds_frag(p, p + 8);
  }
#pragma unroll
  for (int mt = 0; mt < 4; ++mt) {
    const unsigned short* p = sAst + (wm * 64 + mt * 16 + l15) * LROW + hh * 16;
    afr[mt] = lds_frag(p, p + 8);
  }
#pragma unroll
  for (int mt = 0; mt < 4; ++mt)
#pragma unroll
    for (int nt = 0; nt < NT; ++nt)
      acc[mt][nt] = __builtin_amdgcn_wmma_f32_16x16x32_bf16(
          false, afr[mt], false, bfr[nt], (short)0, acc[mt][nt], false, false);
}

template <int NT>
__device__ __forceinline__ void epilogue(const float* __restrict__ bias,
                                         float* __restrict__ out,
                                         int blockRow, int blockCol,
                                         int wm, int wn, int l15, int hh,
                                         v8f (&acc)[4][NT]) {
#pragma unroll
  for (int nt = 0; nt < NT; ++nt) {
    const int col = blockCol + wn * (16 * NT) + nt * 16 + l15;
    const float bv = bias[col];
#pragma unroll
    for (int mt = 0; mt < 4; ++mt) {
      const int rowBase = blockRow + wm * 64 + mt * 16 + hh * 8;
#pragma unroll
      for (int r = 0; r < 8; ++r)
        out[(size_t)(rowBase + r) * DOUT + col] = acc[mt][nt][r] + bv;
    }
  }
}

// ---------------------------------------------------------------------------
// Pre-pass: f32 -> bf16 (RNE), 8 elements/thread, fully coalesced.
// ---------------------------------------------------------------------------
__global__ __launch_bounds__(256)
void cvt_f32_to_bf16(const float* __restrict__ src,
                     unsigned short* __restrict__ dst, int n8) {
  const int i = blockIdx.x * 256 + threadIdx.x;
  if (i >= n8) return;
  const f32x4 a = *(const f32x4*)(src + (size_t)8 * i);
  const f32x4 b = *(const f32x4*)(src + (size_t)8 * i + 4);
  u16x8 u;
#pragma unroll
  for (int j = 0; j < 4; ++j) u[j] = f32_to_bf16_bits(a[j]);
#pragma unroll
  for (int j = 0; j < 4; ++j) u[4 + j] = f32_to_bf16_bits(b[j]);
  *(u16x8*)(dst + (size_t)8 * i) = u;
}

// ---------------------------------------------------------------------------
// Fast GEMM: bf16 sources, 128x256 block, 64x64 per-wave (16 WMMA/stage),
// tile fill entirely via global_load_async_to_lds_b128 (zero VALU).
// ---------------------------------------------------------------------------
__global__ __launch_bounds__(256)
void snn_gemm_bf16src(const unsigned short* __restrict__ Xb,
                      const unsigned short* __restrict__ Wb,
                      const float* __restrict__ bias,
                      float* __restrict__ out) {
  __shared__ unsigned short sA[2][BM * LROW];    // 20 KB
  __shared__ unsigned short sB[2][BNF * LROW];   // 40 KB

  const int tid  = threadIdx.x;
  const int lane = tid & 31;
  const int wid  = tid >> 5;
  const int wm   = wid >> 2, wn = wid & 3;       // 2 x 4 wave grid
  const int l15  = lane & 15, hh = lane >> 4;

  const int blockRow = blockIdx.y * BM;
  const int blockCol = blockIdx.x * BNF;

  // A fill: 2 threads/row, 16 halves each. B fill: 1 thread/row, 32 halves.
  const int fra = tid >> 1;
  const int fsa = (tid & 1) * 16;
  const unsigned short* gA = Xb + (size_t)(blockRow + fra) * DIN + fsa;
  const unsigned short* gB = Wb + (size_t)(blockCol + tid) * DIN;

#if !HAVE_ASYNC_LDS
  u16x8 rga[2], rgb[4];
#endif

  auto fill = [&](int st, int kt) {
    const unsigned short* pa = gA + kt * BK;
    const unsigned short* pb = gB + kt * BK;
    unsigned short* la = &sA[st][fra * LROW + fsa];
    unsigned short* lb = &sB[st][tid * LROW];
#if HAVE_ASYNC_LDS
    async_cp16(pa,      la);
    async_cp16(pa + 8,  la + 8);
    async_cp16(pb,      lb);
    async_cp16(pb + 8,  lb + 8);
    async_cp16(pb + 16, lb + 16);
    async_cp16(pb + 24, lb + 24);
#else
    rga[0] = *(const u16x8*)pa;
    rga[1] = *(const u16x8*)(pa + 8);
#pragma unroll
    for (int i = 0; i < 4; ++i) rgb[i] = *(const u16x8*)(pb + 8 * i);
    *(u16x8*)la       = rga[0];
    *(u16x8*)(la + 8) = rga[1];
#pragma unroll
    for (int i = 0; i < 4; ++i) *(u16x8*)(lb + 8 * i) = rgb[i];
#endif
    if (kt + PFDIST < NSTAGE) {
      __builtin_prefetch(pa + PFDIST * BK, 0, 1);
      __builtin_prefetch(pb + PFDIST * BK, 0, 1);
    }
  };
  auto fill_wait = [&]() {
#if HAVE_ASYNC_LDS
    asm volatile("s_wait_asynccnt 0x0" ::: "memory");
#endif
  };

  v8f acc[4][4] = {};

  fill(0, 0);
  fill_wait();
  __syncthreads();
  for (int kt = 0; kt < NSTAGE; ++kt) {
    if (kt + 1 < NSTAGE) fill((kt + 1) & 1, kt + 1);
    mma_stage<4>(&sA[kt & 1][0], &sB[kt & 1][0], wm, wn, l15, hh, acc);
    if (kt + 1 < NSTAGE) fill_wait();
    __syncthreads();
  }

  epilogue<4>(bias, out, blockRow, blockCol, wm, wn, l15, hh, acc);
}

// ---------------------------------------------------------------------------
// Fallback GEMM (small ws): f32 sources, 128x128 block, convert during fill.
// ---------------------------------------------------------------------------
__global__ __launch_bounds__(256)
void snn_gemm_f32src(const float* __restrict__ X, const float* __restrict__ W,
                     const float* __restrict__ bias, float* __restrict__ out) {
  __shared__ unsigned short sA[2][BM * LROW];
  __shared__ unsigned short sB[2][BNS * LROW];

  const int tid  = threadIdx.x;
  const int lane = tid & 31;
  const int wid  = tid >> 5;
  const int wm   = wid >> 2, wn = wid & 3;
  const int l15  = lane & 15, hh = lane >> 4;

  const int blockRow = blockIdx.y * BM;
  const int blockCol = blockIdx.x * BNS;

  const int fr = tid >> 1;
  const int fs = (tid & 1) * 16;
  const float* gA = X + (size_t)(blockRow + fr) * DIN + fs;
  const float* gB = W + (size_t)(blockCol + fr) * DIN + fs;

  f32x4 ra[4], rb[4];

  auto loadG = [&](int kt) {
    const float* pa = gA + kt * BK;
    const float* pb = gB + kt * BK;
#pragma unroll
    for (int i = 0; i < 4; ++i) {
      ra[i] = *(const f32x4*)(pa + 4 * i);
      rb[i] = *(const f32x4*)(pb + 4 * i);
    }
    if (kt + PFDIST < NSTAGE) {
      __builtin_prefetch(pa + PFDIST * BK, 0, 1);
      __builtin_prefetch(pb + PFDIST * BK, 0, 1);
    }
  };
  auto storeL = [&](int st) {
#pragma unroll
    for (int i = 0; i < 2; ++i) {
      u16x8 va, vb;
#pragma unroll
      for (int j = 0; j < 8; ++j) {
        va[j] = f32_to_bf16_bits(ra[2 * i + (j >> 2)][j & 3]);
        vb[j] = f32_to_bf16_bits(rb[2 * i + (j >> 2)][j & 3]);
      }
      *(u16x8*)(&sA[st][fr * LROW + fs + 8 * i]) = va;
      *(u16x8*)(&sB[st][fr * LROW + fs + 8 * i]) = vb;
    }
  };

  v8f acc[4][2] = {};

  loadG(0);
  storeL(0);
  __syncthreads();
  for (int kt = 0; kt < NSTAGE; ++kt) {
    if (kt + 1 < NSTAGE) loadG(kt + 1);
    mma_stage<2>(&sA[kt & 1][0], &sB[kt & 1][0], wm, wn, l15, hh, acc);
    if (kt + 1 < NSTAGE) storeL((kt + 1) & 1);
    __syncthreads();
  }

  epilogue<2>(bias, out, blockRow, blockCol, wm, wn, l15, hh, acc);
}

// ---------------------------------------------------------------------------
// LIF scan in place on the GEMM result (spikes overwrite currents).
// ---------------------------------------------------------------------------
__global__ __launch_bounds__(256)
void lif_scan(float* __restrict__ cur) {
  const int idx = blockIdx.x * 256 + threadIdx.x;   // 0 .. B*DOUT-1
  const int b = idx >> 11;
  const int e = idx & (DOUT - 1);
  const size_t base = (size_t)b * TSTEPS * DOUT + e;
  float v = 0.0f;
#pragma unroll
  for (int t = 0; t < TSTEPS; ++t) {
    const size_t off = base + (size_t)t * DOUT;
    const float c = cur[off];
    v = v * TAU + c;
    const float s = (v > VTH) ? 1.0f : 0.0f;
    cur[off] = s;
    v = v * (1.0f - s);
  }
}

extern "C" void kernel_launch(void* const* d_in, const int* in_sizes, int n_in,
                              void* d_out, int out_size, void* d_ws, size_t ws_size,
                              hipStream_t stream) {
  const float* x  = (const float*)d_in[0];  // [512, 20, 2048] f32
  const float* W  = (const float*)d_in[1];  // [2048, 2048] f32
  const float* bv = (const float*)d_in[2];  // [2048] f32
  float* out = (float*)d_out;               // [512, 20, 2048] f32

  if (ws_size >= WS_NEEDED) {
    unsigned short* xb = (unsigned short*)d_ws;
    unsigned short* wb = xb + XELEMS;
    cvt_f32_to_bf16<<<(int)(XELEMS / 8 / 256), 256, 0, stream>>>(x, xb, (int)(XELEMS / 8));
    cvt_f32_to_bf16<<<(int)(WELEMS / 8 / 256), 256, 0, stream>>>(W, wb, (int)(WELEMS / 8));
    const dim3 g1(DOUT / BNF, MROWS / BM);  // (8, 80)
    snn_gemm_bf16src<<<g1, 256, 0, stream>>>(xb, wb, bv, out);
  } else {
    const dim3 g1(DOUT / BNS, MROWS / BM);  // (16, 80)
    snn_gemm_f32src<<<g1, 256, 0, stream>>>(x, W, bv, out);
  }

  const int nScan = (NBATCH * DOUT) / 256;  // 4096
  lif_scan<<<nScan, 256, 0, stream>>>(out);
}
